// SOMPCell_10041633538618
// MI455X (gfx1250) — compile-verified, matrix operand order checked by reading, not run on hardware
//
#include <hip/hip_runtime.h>
#include <math.h>

// ---------------------------------------------------------------------------
// MI455X (gfx1250) fused dual-GEMM + epilogue, wave32, WMMA bf16x3.
//   acc = x1@W1 + x2@W2   (f32 accum, 3x v_wmma_f32_16x16x32_bf16 per tile)
//   Y   = tanh(0.9*bucket + 0.5*acc + 0.5*(b1+b2) + b + 0.1*noise - 0.1*|prev|)
//
// Fast path (needs ~96MB of d_ws):
//   prep kernel:  W1,W2 -> transposed bf16 hi/lo planes in d_ws (done once per
//                 launch; weights are reused by 64 M-tile blocks, so the split
//                 + transpose is hoisted out of the GEMM hot loop)
//   gemm kernel:  B tiles DMA'd global->LDS with GLOBAL_LOAD_ASYNC_TO_LDS_B128
//                 (ASYNCcnt), double-buffered dynamic LDS, one barrier/step.
// Fallback path: round-1 kernel (everything converted in-kernel).
// ---------------------------------------------------------------------------

typedef __attribute__((ext_vector_type(16))) __bf16 v16bf;
typedef __attribute__((ext_vector_type(8)))  __bf16 v8bf;
typedef __attribute__((ext_vector_type(4)))  __bf16 v4bf;
typedef __attribute__((ext_vector_type(8)))  float  v8f;

#define M_TOT    8192      // B*S
#define N_TOT    4096      // DO
#define K1       4096      // D1
#define K2       2048      // D2
#define BM       128
#define BN       128
#define BK       32
#define LDT      40        // padded LDS row stride (elements): 80B, 16B-aligned
#define NTHREADS 256       // 8 wave32 waves
#define NSTEPS   ((K1 / BK) + (K2 / BK))   // 192

// Dynamic-LDS layout (bytes), double buffered
#define SA_HI_OFF 0u
#define SA_LO_OFF 10240u
#define SB_HI_OFF 20480u
#define SB_LO_OFF 30720u
#define BUF_SZ    40960u
#define LDS_TOTAL (2u * BUF_SZ)            // 80 KB

// 16-element bf16 fragment row: halves at +kb and +kb+16
// (A-matrix 16-bit K striping per CDNA5 ISA 7.12.2; B assumed symmetric).
__device__ __forceinline__ v16bf frag16(const __bf16* rowbase, int kb) {
  v8bf p = *(const v8bf*)(rowbase + kb);
  v8bf q = *(const v8bf*)(rowbase + kb + 16);
  return __builtin_shufflevector(p, q,
      0, 1, 2, 3, 4, 5, 6, 7, 8, 9, 10, 11, 12, 13, 14, 15);
}

// Async DMA: 16 bytes global -> LDS, tracked by ASYNCcnt (gfx1250).
__device__ __forceinline__ void async_b128(unsigned lds_addr, const void* gsrc) {
  asm volatile("global_load_async_to_lds_b128 %0, %1, off"
               :: "v"(lds_addr), "v"(gsrc) : "memory");
}
__device__ __forceinline__ void wait_async0() {
  asm volatile("s_wait_asynccnt 0x0" ::: "memory");
}

// ---------------------------------------------------------------------------
// Prep: W (K x N, f32) -> transposed bf16 hi/lo planes Wt (N x K). 32x32 tiles.
// ---------------------------------------------------------------------------
__global__ __launch_bounds__(256)
void transpose_split_bf16(const float* __restrict__ W,
                          __bf16* __restrict__ th, __bf16* __restrict__ tl,
                          int K, int N) {
  __shared__ float tile[32][33];
  const int n0 = blockIdx.x * 32, k0 = blockIdx.y * 32;
  const int tx = threadIdx.x, ty = threadIdx.y;   // 32 x 8
#pragma unroll
  for (int j = 0; j < 4; ++j) {
    int k = ty + j * 8;
    tile[k][tx] = W[(size_t)(k0 + k) * N + n0 + tx];
  }
  __syncthreads();
#pragma unroll
  for (int j = 0; j < 4; ++j) {
    int    n = ty + j * 8;
    float  x = tile[tx][n];                       // W[k0+tx][n0+n]
    __bf16 h = (__bf16)x;
    __bf16 l = (__bf16)(x - (float)h);
    size_t o = (size_t)(n0 + n) * K + k0 + tx;    // Wt[n][k], coalesced in k
    th[o] = h;
    tl[o] = l;
  }
}

// ---------------------------------------------------------------------------
// Fast GEMM: pre-split transposed weights in ws, async B-tile DMA to LDS.
// ---------------------------------------------------------------------------
__global__ __launch_bounds__(NTHREADS)
void somp_gemm_async(const float* __restrict__ x1, const float* __restrict__ x2,
                     const float* __restrict__ noise,
                     const __bf16* __restrict__ w1th, const __bf16* __restrict__ w1tl,
                     const __bf16* __restrict__ w2th, const __bf16* __restrict__ w2tl,
                     const float* __restrict__ b1, const float* __restrict__ b2,
                     const float* __restrict__ bvec,
                     const float* __restrict__ bucket,
                     const float* __restrict__ prev,
                     float* __restrict__ out) {
  extern __shared__ __align__(16) unsigned char smem[];

  const int tid  = threadIdx.x;
  const int lane = tid & 31;
  const int wv   = tid >> 5;
  const int wm   = wv & 3;            // 4 waves along M -> 32 rows each
  const int wn   = wv >> 2;           // 2 waves along N -> 64 cols each
  const int m0   = blockIdx.y * BM;
  const int n0   = blockIdx.x * BN;
  const int ln   = lane & 15;
  const int kb   = (lane >> 4) << 3;  // K fragment base: 0 or 8
  const int mofs = (lane >> 4) << 3;  // C layout M offset: 0 or 8
  const unsigned lds0 = (unsigned)(uintptr_t)smem;  // LDS byte offset of base

  v8f acc[2][4];
#pragma unroll
  for (int f = 0; f < 2; ++f)
#pragma unroll
    for (int g = 0; g < 4; ++g) acc[f][g] = {};

  float4 areg[4];

  // ---- A tile: fp32 global -> regs ----
  auto load_a = [&](int s) {
    const float* A;
    int ldk, k0;
    if (s < K1 / BK) { A = x1; ldk = K1; k0 = s * BK; }
    else             { A = x2; ldk = K2; k0 = (s - K1 / BK) * BK; }
#pragma unroll
    for (int i = 0; i < 4; ++i) {
      int q   = tid + i * NTHREADS;   // 0..1023 float4 slots (128 rows x 8)
      int row = q >> 3, c4 = q & 7;
      areg[i] = *(const float4*)(A + (size_t)(m0 + row) * ldk + k0 + c4 * 4);
    }
  };

  // ---- A tile: split to bf16 hi/lo, packed ds_store_b64 into LDS[buf] ----
  auto store_a = [&](unsigned base) {
    __bf16* sahi = (__bf16*)(smem + base + SA_HI_OFF);
    __bf16* salo = (__bf16*)(smem + base + SA_LO_OFF);
#pragma unroll
    for (int i = 0; i < 4; ++i) {
      int q   = tid + i * NTHREADS;
      int row = q >> 3, c4 = q & 7;
      const float* av = (const float*)&areg[i];
      v4bf h4, l4;
#pragma unroll
      for (int j = 0; j < 4; ++j) {
        float x = av[j];
        h4[j]   = (__bf16)x;
        l4[j]   = (__bf16)(x - (float)h4[j]);
      }
      *(v4bf*)(&sahi[row * LDT + c4 * 4]) = h4;
      *(v4bf*)(&salo[row * LDT + c4 * 4]) = l4;
    }
  };

  // ---- B tile: async DMA bf16 planes (pre-transposed [n][k]) -> LDS[buf] --
  auto issue_b = [&](int s, unsigned base) {
    const __bf16 *wth, *wtl;
    int K, k0;
    if (s < K1 / BK) { wth = w1th; wtl = w1tl; K = K1; k0 = s * BK; }
    else             { wth = w2th; wtl = w2tl; K = K2; k0 = (s - K1 / BK) * BK; }
#pragma unroll
    for (int i = 0; i < 2; ++i) {
      int      c  = tid + i * NTHREADS;   // 512 chunks: 128 rows x 4 x 16B
      int      n  = c >> 2, q = c & 3;
      size_t   go = (size_t)(n0 + n) * K + k0 + q * 8;
      unsigned lo = (unsigned)(n * LDT + q * 8) * 2u;
      async_b128(lds0 + base + SB_HI_OFF + lo, wth + go);
      async_b128(lds0 + base + SB_LO_OFF + lo, wtl + go);
    }
  };

  // ---- pipeline: 1 barrier/step, async B prefetch in flight during WMMA ---
  issue_b(0, 0);
  load_a(0);
  unsigned base = 0;
  for (int s = 0; s < NSTEPS; ++s) {
    store_a(base);
    if (s + 1 < NSTEPS) load_a(s + 1);
    wait_async0();                       // B(s) landed in LDS (per-wave)
    __syncthreads();                     // publish A stores + close buf^1 reads
    if (s + 1 < NSTEPS) issue_b(s + 1, base ^ BUF_SZ);

    const __bf16* sahi = (const __bf16*)(smem + base + SA_HI_OFF);
    const __bf16* salo = (const __bf16*)(smem + base + SA_LO_OFF);
    const __bf16* sbhi = (const __bf16*)(smem + base + SB_HI_OFF);
    const __bf16* sblo = (const __bf16*)(smem + base + SB_LO_OFF);

    v16bf ah[2], al[2], bh[4], bl[4];
#pragma unroll
    for (int f = 0; f < 2; ++f) {
      int row = wm * 32 + f * 16 + ln;
      ah[f] = frag16(&sahi[row * LDT], kb);
      al[f] = frag16(&salo[row * LDT], kb);
    }
#pragma unroll
    for (int g = 0; g < 4; ++g) {
      int col = wn * 64 + g * 16 + ln;
      bh[g] = frag16(&sbhi[col * LDT], kb);
      bl[g] = frag16(&sblo[col * LDT], kb);
    }

#pragma unroll
    for (int f = 0; f < 2; ++f)
#pragma unroll
      for (int g = 0; g < 4; ++g) {
        // bf16x3: hi*hi + hi*lo + lo*hi (~fp32 accuracy, f32 accumulate)
        acc[f][g] = __builtin_amdgcn_wmma_f32_16x16x32_bf16(
            false, ah[f], false, bh[g], (short)0, acc[f][g], false, false);
        acc[f][g] = __builtin_amdgcn_wmma_f32_16x16x32_bf16(
            false, ah[f], false, bl[g], (short)0, acc[f][g], false, false);
        acc[f][g] = __builtin_amdgcn_wmma_f32_16x16x32_bf16(
            false, al[f], false, bh[g], (short)0, acc[f][g], false, false);
      }
    base ^= BUF_SZ;
  }

  // ---- fused epilogue (C layout: lane%16 = N, VGPR r = M row, +8 hi half) --
#pragma unroll
  for (int g = 0; g < 4; ++g) {
    int   gn    = n0 + wn * 64 + g * 16 + ln;
    float extra = 0.5f * (b1[gn] + b2[gn]) + bvec[gn] +
                  0.9f * bucket[gn] - 0.1f * fabsf(prev[gn]);
#pragma unroll
    for (int f = 0; f < 2; ++f) {
      int gmb = m0 + wm * 32 + f * 16 + mofs;
#pragma unroll
      for (int r = 0; r < 8; ++r) {
        size_t idx = (size_t)(gmb + r) * N_TOT + gn;
        float  nz  = __builtin_nontemporal_load(&noise[idx]);
        float  Z   = 0.5f * acc[f][g][r] + 0.1f * nz + extra;
        __builtin_nontemporal_store(tanhf(Z), &out[idx]);
      }
    }
  }
}

// ---------------------------------------------------------------------------
// Fallback (round-1): everything converted in-kernel, no workspace needed.
// ---------------------------------------------------------------------------
__global__ __launch_bounds__(NTHREADS)
void somp_gemm_fallback(const float* __restrict__ x1, const float* __restrict__ x2,
                        const float* __restrict__ noise,
                        const float* __restrict__ W1, const float* __restrict__ b1,
                        const float* __restrict__ W2, const float* __restrict__ b2,
                        const float* __restrict__ bvec,
                        const float* __restrict__ bucket,
                        const float* __restrict__ prev,
                        float* __restrict__ out) {
  __shared__ __align__(16) __bf16 sa_hi[BM * LDT];
  __shared__ __align__(16) __bf16 sa_lo[BM * LDT];
  __shared__ __align__(16) __bf16 sb_hi[BN * LDT];
  __shared__ __align__(16) __bf16 sb_lo[BN * LDT];

  const int tid  = threadIdx.x;
  const int lane = tid & 31;
  const int wv   = tid >> 5;
  const int wm   = wv & 3;
  const int wn   = wv >> 2;
  const int m0   = blockIdx.y * BM;
  const int n0   = blockIdx.x * BN;
  const int ln   = lane & 15;
  const int kb   = (lane >> 4) << 3;
  const int mofs = (lane >> 4) << 3;

  v8f acc[2][4];
#pragma unroll
  for (int f = 0; f < 2; ++f)
#pragma unroll
    for (int g = 0; g < 4; ++g) acc[f][g] = {};

  float4 areg[4], breg[4];

  auto load_step = [&](int s) {
    const float* A;
    const float* W;
    int ldk, k0;
    if (s < K1 / BK) { A = x1; W = W1; ldk = K1; k0 = s * BK; }
    else             { A = x2; W = W2; ldk = K2; k0 = (s - K1 / BK) * BK; }
#pragma unroll
    for (int i = 0; i < 4; ++i) {
      int q   = tid + i * NTHREADS;
      int row = q >> 3, c4 = q & 7;
      areg[i] = *(const float4*)(A + (size_t)(m0 + row) * ldk + k0 + c4 * 4);
      int kr = q >> 5, cb = q & 31;
      breg[i] = *(const float4*)(W + (size_t)(k0 + kr) * N_TOT + n0 + cb * 4);
    }
  };

  auto store_step = [&]() {
#pragma unroll
    for (int i = 0; i < 4; ++i) {
      int q   = tid + i * NTHREADS;
      int row = q >> 3, c4 = q & 7;
      const float* av = (const float*)&areg[i];
      v4bf h4, l4;
#pragma unroll
      for (int j = 0; j < 4; ++j) {
        float x = av[j];
        h4[j]   = (__bf16)x;
        l4[j]   = (__bf16)(x - (float)h4[j]);
      }
      *(v4bf*)(&sa_hi[row * LDT + c4 * 4]) = h4;
      *(v4bf*)(&sa_lo[row * LDT + c4 * 4]) = l4;
      int kr = q >> 5, cb = q & 31;
      const float* bv = (const float*)&breg[i];
#pragma unroll
      for (int j = 0; j < 4; ++j) {
        float  x = bv[j];
        __bf16 h = (__bf16)x;
        __bf16 l = (__bf16)(x - (float)h);
        int    n = cb * 4 + j;
        sb_hi[n * LDT + kr] = h;
        sb_lo[n * LDT + kr] = l;
      }
    }
  };

  load_step(0);
  for (int s = 0; s < NSTEPS; ++s) {
    if (s) __syncthreads();
    store_step();
    __syncthreads();
    if (s + 1 < NSTEPS) load_step(s + 1);

    v16bf ah[2], al[2], bh[4], bl[4];
#pragma unroll
    for (int f = 0; f < 2; ++f) {
      int row = wm * 32 + f * 16 + ln;
      ah[f] = frag16(&sa_hi[row * LDT], kb);
      al[f] = frag16(&sa_lo[row * LDT], kb);
    }
#pragma unroll
    for (int g = 0; g < 4; ++g) {
      int col = wn * 64 + g * 16 + ln;
      bh[g] = frag16(&sb_hi[col * LDT], kb);
      bl[g] = frag16(&sb_lo[col * LDT], kb);
    }
#pragma unroll
    for (int f = 0; f < 2; ++f)
#pragma unroll
      for (int g = 0; g < 4; ++g) {
        acc[f][g] = __builtin_amdgcn_wmma_f32_16x16x32_bf16(
            false, ah[f], false, bh[g], (short)0, acc[f][g], false, false);
        acc[f][g] = __builtin_amdgcn_wmma_f32_16x16x32_bf16(
            false, ah[f], false, bl[g], (short)0, acc[f][g], false, false);
        acc[f][g] = __builtin_amdgcn_wmma_f32_16x16x32_bf16(
            false, al[f], false, bh[g], (short)0, acc[f][g], false, false);
      }
  }

#pragma unroll
  for (int g = 0; g < 4; ++g) {
    int   gn    = n0 + wn * 64 + g * 16 + ln;
    float extra = 0.5f * (b1[gn] + b2[gn]) + bvec[gn] +
                  0.9f * bucket[gn] - 0.1f * fabsf(prev[gn]);
#pragma unroll
    for (int f = 0; f < 2; ++f) {
      int gmb = m0 + wm * 32 + f * 16 + mofs;
#pragma unroll
      for (int r = 0; r < 8; ++r) {
        size_t idx = (size_t)(gmb + r) * N_TOT + gn;
        float  nz  = __builtin_nontemporal_load(&noise[idx]);
        float  Z   = 0.5f * acc[f][g][r] + 0.1f * nz + extra;
        __builtin_nontemporal_store(tanhf(Z), &out[idx]);
      }
    }
  }
}

// ---------------------------------------------------------------------------
extern "C" void kernel_launch(void* const* d_in, const int* in_sizes, int n_in,
                              void* d_out, int out_size, void* d_ws,
                              size_t ws_size, hipStream_t stream) {
  (void)in_sizes; (void)n_in; (void)out_size;
  const float* x1     = (const float*)d_in[0];
  const float* x2     = (const float*)d_in[1];
  const float* noise  = (const float*)d_in[2];
  const float* W1     = (const float*)d_in[3];
  const float* b1     = (const float*)d_in[4];
  const float* W2     = (const float*)d_in[5];
  const float* b2     = (const float*)d_in[6];
  const float* bvec   = (const float*)d_in[7];
  const float* bucket = (const float*)d_in[8];
  const float* prev   = (const float*)d_in[9];
  float*       out    = (float*)d_out;

  const size_t S1      = (size_t)N_TOT * K1;          // W1t plane elements
  const size_t S2      = (size_t)N_TOT * K2;          // W2t plane elements
  const size_t WS_NEED = (2 * S1 + 2 * S2) * sizeof(__bf16);  // ~96 MB

  dim3 grid(N_TOT / BN, M_TOT / BM);                  // 32 x 64 workgroups

  if (ws_size >= WS_NEED) {
    __bf16* w1th = (__bf16*)d_ws;
    __bf16* w1tl = w1th + S1;
    __bf16* w2th = w1tl + S1;
    __bf16* w2tl = w2th + S2;

    dim3 pb(32, 8);
    transpose_split_bf16<<<dim3(N_TOT / 32, K1 / 32), pb, 0, stream>>>(
        W1, w1th, w1tl, K1, N_TOT);
    transpose_split_bf16<<<dim3(N_TOT / 32, K2 / 32), pb, 0, stream>>>(
        W2, w2th, w2tl, K2, N_TOT);

    somp_gemm_async<<<grid, NTHREADS, LDS_TOTAL, stream>>>(
        x1, x2, noise, w1th, w1tl, w2th, w2tl, b1, b2, bvec, bucket, prev, out);
  } else {
    somp_gemm_fallback<<<grid, NTHREADS, 0, stream>>>(
        x1, x2, noise, W1, b1, W2, b2, bvec, bucket, prev, out);
  }
}
